// FDTD2DLayer_17978733101082
// MI455X (gfx1250) — matrix operand order repaired and checked by reading, not compiled, not run on hardware
//
#include <hip/hip_runtime.h>
#include <hip/hip_bf16.h>

// ---------------------------------------------------------------------------
// FDTD2D layer for gfx1250 (MI455X): two bf16 WMMA GEMMs + chunked complex scan
//  - GEMM K-loops software-pipelined (double-buffered fragments) so v_wmma
//    overlaps the next K-slab's global_load_b128 clauses.
//  - Scan pass 1 fused into GEMM1 epilogue via a 128x64 LDS tile (the GEMM1
//    block tile IS one scan chunk), saving a full 33.5 MB re-read of U.
// ---------------------------------------------------------------------------

#define SEQ   8192
#define Hdim  512
#define Gdim  1024
#define CHUNK 128           // scan chunk length == GEMM1 block M-tile
#define NCH   (SEQ / CHUNK) // 64 chunks

typedef __attribute__((ext_vector_type(16))) __bf16 v16bf;
typedef __attribute__((ext_vector_type(8)))  __bf16 v8bf;
typedef __attribute__((ext_vector_type(8)))  float  v8f;

// ---------------- helpers ----------------

__device__ __forceinline__ unsigned short f32_to_bf16_rne(float f) {
    union { float f; unsigned int u; } v; v.f = f;
    unsigned int u = v.u;
    unsigned int r = u + 0x7fffu + ((u >> 16) & 1u);
    return (unsigned short)(r >> 16);
}

__device__ __forceinline__ float sp(float x) { return log1pf(expf(x)); } // softplus
__device__ __forceinline__ float clipf(float x, float lo, float hi) {
    return fminf(fmaxf(x, lo), hi);
}

// A fragment: 16x32 bf16, row-major source, lda elements per row.
// lane l: m = l&15, kb = (l>>4)*8 ; chunks [k0+kb, +8) and [k0+kb+16, +8)
__device__ __forceinline__ v16bf load_a_frag(const __bf16* __restrict__ A, int lda,
                                             int mtile, int k0, int lane) {
    int m  = mtile + (lane & 15);
    int kb = (lane >> 4) * 8;
    const __bf16* row = A + (size_t)m * lda + k0 + kb;
    v8bf lo = *(const v8bf*)(const void*)(row);
    v8bf hi = *(const v8bf*)(const void*)(row + 16);
    return __builtin_shufflevector(lo, hi, 0,1,2,3,4,5,6,7,8,9,10,11,12,13,14,15);
}

// B fragment: 32x16 bf16 (K x N).  Source stored (N,K) row-major, ldb per row.
// lane l: n = l&15, kb = (l>>4)*8 ; chunks [k0+kb, +8) and [k0+kb+16, +8)
__device__ __forceinline__ v16bf load_b_frag(const __bf16* __restrict__ Bm, int ldb,
                                             int ntile, int k0, int lane) {
    int n  = ntile + (lane & 15);
    int kb = (lane >> 4) * 8;
    const __bf16* row = Bm + (size_t)n * ldb + k0 + kb;
    v8bf lo = *(const v8bf*)(const void*)(row);
    v8bf hi = *(const v8bf*)(const void*)(row + 16);
    return __builtin_shufflevector(lo, hi, 0,1,2,3,4,5,6,7,8,9,10,11,12,13,14,15);
}

#define WMMA_BF16(a, b, c) \
    __builtin_amdgcn_wmma_f32_16x16x32_bf16(false, (a), false, (b), (short)0, (c), false, false)

// ---------------- kernels ----------------

// f32 -> bf16 (as ushort bits), 4 elements per thread, n multiple of 4
__global__ void cvt_bf16_kernel(const float* __restrict__ in,
                                unsigned short* __restrict__ out, int n) {
    int i = (blockIdx.x * blockDim.x + threadIdx.x) * 4;
    if (i >= n) return;
    float4 f = *(const float4*)(const void*)(in + i);
    unsigned long long p =
        (unsigned long long)f32_to_bf16_rne(f.x)
      | ((unsigned long long)f32_to_bf16_rne(f.y) << 16)
      | ((unsigned long long)f32_to_bf16_rne(f.z) << 32)
      | ((unsigned long long)f32_to_bf16_rne(f.w) << 48);
    *(unsigned long long*)(void*)(out + i) = p;
}

// Per-channel params: scaling[g], lam2 = lr[g] + i*li[g]   (G = 1024 threads)
__global__ void params_kernel(const float* __restrict__ c,
                              const float* __restrict__ kp_diag,
                              const float* __restrict__ k_diag,
                              const float* __restrict__ dtm,
                              float* __restrict__ scaling,
                              float* __restrict__ lr,
                              float* __restrict__ li) {
    int g = blockIdx.x * blockDim.x + threadIdx.x;
    if (g >= Gdim) return;
    float dt   = clipf(expf(dtm[g]), 0.1f, 5.0f);            // dt2d, also dt_s
    float maxc = 0.7f / dt;                                   // dx = 1
    float c_pos = clipf(sp(c[g]), 0.01f, 0.9f * maxc);

    float kpf = 0.0f, kf = 0.0f;
    if ((g % 33) == 0) {                                      // idx = arange(0,G,33)[:32]
        int i = g / 33;
        float kp_d = clipf(sp(kp_diag[i]), 1e-4f, 0.2f);
        float k_d  = clipf(sp(k_diag[i]),  1e-4f, 0.2f);
        kpf = clipf(sp(kp_d), 1e-4f, 0.5f);                   // kp_d2
        kf  = clipf(sp(k_d),  1e-4f, 0.5f);                   // k_d2
    }
    kpf = clipf(sp(kpf), 1e-4f, 0.5f);                        // saturates to 0.5
    kf  = clipf(sp(kf),  1e-4f, 0.5f);

    float d1 = fmaxf(1.0f + dt * kf,  0.1f);
    float d2 = fmaxf(1.0f + dt * kpf, 0.1f);
    float realp  = 0.5f * (1.0f / d2 + 1.0f / d1);
    float c_safe = clipf(c_pos, 0.01f, 1.0f);
    const float xi = 1.5707963267948966f;                     // pi/(2*dx)
    float imagp = clipf(c_safe * dt * xi * rsqrtf(fmaxf(d1 * d2, 1e-6f)), -10.0f, 10.0f);

    scaling[g] = 1.0f / (1.0f + dt * kpf);
    lr[g] = realp;
    li[g] = imagp;
}

// GEMM1 + fused scan pass 1.
// U[t][g] = (sum_h x[t][h] * B[g][h]) * scaling[g]
// M=SEQ, N=Gdim, K=Hdim.  Block: 8 waves, tile 128(M) x 64(N) == one scan chunk.
// Epilogue stages the tile in LDS; 64 threads then run the chunk-local complex
// recurrence down their column and emit the chunk-end carries Ea/Eb.
__global__ void __launch_bounds__(256)
gemm1_kernel(const __bf16* __restrict__ Xb,   // SEQ x H   (row-major)
             const __bf16* __restrict__ Bb,   // G x H     (row-major) == B-operand (N,K)
             const float*  __restrict__ scaling,
             const float*  __restrict__ lr,
             const float*  __restrict__ li,
             float* __restrict__ U,           // SEQ x G
             float* __restrict__ Ea,          // NCH x G
             float* __restrict__ Eb) {
    __shared__ float tile[CHUNK * 64];        // 32 KB

    const int lane  = threadIdx.x & 31;
    const int wave  = threadIdx.x >> 5;
    const int mtile = blockIdx.y * 128 + wave * 16;
    const int ntile = blockIdx.x * 64;

    v8f acc0 = {}, acc1 = {}, acc2 = {}, acc3 = {};

    // prologue: fragments for k0 = 0
    v16bf a  = load_a_frag(Xb, Hdim, mtile, 0, lane);
    v16bf b0 = load_b_frag(Bb, Hdim, ntile +  0, 0, lane);
    v16bf b1 = load_b_frag(Bb, Hdim, ntile + 16, 0, lane);
    v16bf b2 = load_b_frag(Bb, Hdim, ntile + 32, 0, lane);
    v16bf b3 = load_b_frag(Bb, Hdim, ntile + 48, 0, lane);

#pragma unroll 2
    for (int k0 = 0; k0 < Hdim - 32; k0 += 32) {
        // issue next K-slab loads, then consume current fragments
        v16bf an  = load_a_frag(Xb, Hdim, mtile, k0 + 32, lane);
        v16bf bn0 = load_b_frag(Bb, Hdim, ntile +  0, k0 + 32, lane);
        v16bf bn1 = load_b_frag(Bb, Hdim, ntile + 16, k0 + 32, lane);
        v16bf bn2 = load_b_frag(Bb, Hdim, ntile + 32, k0 + 32, lane);
        v16bf bn3 = load_b_frag(Bb, Hdim, ntile + 48, k0 + 32, lane);
        acc0 = WMMA_BF16(a, b0, acc0);
        acc1 = WMMA_BF16(a, b1, acc1);
        acc2 = WMMA_BF16(a, b2, acc2);
        acc3 = WMMA_BF16(a, b3, acc3);
        a = an; b0 = bn0; b1 = bn1; b2 = bn2; b3 = bn3;
    }
    acc0 = WMMA_BF16(a, b0, acc0);
    acc1 = WMMA_BF16(a, b1, acc1);
    acc2 = WMMA_BF16(a, b2, acc2);
    acc3 = WMMA_BF16(a, b3, acc3);

    // epilogue: scale, write U (global) and the LDS tile for the fused scan
    const int ncol  = lane & 15;
    const int lrow0 = wave * 16 + (lane >> 4) * 8;            // row within block tile
#pragma unroll
    for (int j = 0; j < 8; ++j) {
        size_t r  = (size_t)(mtile + (lane >> 4) * 8 + j) * Gdim + ntile;
        int    lr_ = (lrow0 + j) * 64;
        float v0 = acc0[j] * scaling[ntile + ncol +  0];
        float v1 = acc1[j] * scaling[ntile + ncol + 16];
        float v2 = acc2[j] * scaling[ntile + ncol + 32];
        float v3 = acc3[j] * scaling[ntile + ncol + 48];
        U[r + ncol +  0] = v0;  tile[lr_ + ncol +  0] = v0;
        U[r + ncol + 16] = v1;  tile[lr_ + ncol + 16] = v1;
        U[r + ncol + 32] = v2;  tile[lr_ + ncol + 32] = v2;
        U[r + ncol + 48] = v3;  tile[lr_ + ncol + 48] = v3;
    }
    __syncthreads();

    // fused scan pass 1: 64 threads, one column each (column == LDS bank -> no
    // conflicts).  chunk id == blockIdx.y.
    if (threadIdx.x < 64) {
        int   col = threadIdx.x;
        int   g   = ntile + col;
        float r   = lr[g], im = li[g];
        float sa = 0.0f, sb = 0.0f;
#pragma unroll 4
        for (int t = 0; t < CHUNK; ++t) {
            float un = tile[t * 64 + col];
            float na = fmaf(r, sa, fmaf(-im, sb, un));
            float nb = fmaf(im, sa, r * sb);
            sa = na; sb = nb;
        }
        Ea[blockIdx.y * Gdim + g] = sa;
        Eb[blockIdx.y * Gdim + g] = sb;
    }
}

// Scan pass 2: propagate carries across chunks: S_c = lam^CHUNK * S_{c-1} + E_{c-1}
__global__ void scan_pass2(const float* __restrict__ lr,
                           const float* __restrict__ li,
                           const float* __restrict__ Ea,
                           const float* __restrict__ Eb,
                           float* __restrict__ Sa, float* __restrict__ Sb) {
    int g = blockIdx.x * blockDim.x + threadIdx.x;
    if (g >= Gdim) return;
    float pr = lr[g], pi = li[g];
#pragma unroll
    for (int s = 0; s < 7; ++s) {                    // lam^128 via squaring
        float nr = pr * pr - pi * pi;
        float ni = 2.0f * pr * pi;
        pr = nr; pi = ni;
    }
    float a = 0.0f, b = 0.0f;
    for (int c = 0; c < NCH; ++c) {
        Sa[c * Gdim + g] = a;
        Sb[c * Gdim + g] = b;
        float ea = Ea[c * Gdim + g], eb = Eb[c * Gdim + g];
        float na = fmaf(pr, a, fmaf(-pi, b, ea));
        float nb = fmaf(pi, a, fmaf(pr, b, eb));
        a = na; b = nb;
    }
}

// Scan pass 3: replay with correct initial state, emit P = bf16(2*Re(y))
__global__ void scan_pass3(const float* __restrict__ U,
                           const float* __restrict__ lr,
                           const float* __restrict__ li,
                           const float* __restrict__ Sa,
                           const float* __restrict__ Sb,
                           unsigned short* __restrict__ P) {
    int g = blockIdx.x * blockDim.x + threadIdx.x;
    int c = blockIdx.y;
    float r = lr[g], im = li[g];
    float a = Sa[c * Gdim + g], b = Sb[c * Gdim + g];
    const float* u = U + (size_t)c * CHUNK * Gdim + g;
    unsigned short* p = P + (size_t)c * CHUNK * Gdim + g;
#pragma unroll 4
    for (int t = 0; t < CHUNK; ++t) {
        float un = u[(size_t)t * Gdim];
        float na = fmaf(r, a, fmaf(-im, b, un));
        float nb = fmaf(im, a, r * b);
        a = na; b = nb;
        p[(size_t)t * Gdim] = f32_to_bf16_rne(2.0f * a);
    }
}

// GEMM2: out[t][j] = sum_g P[t][g]*C[j][g] + x[t][j]*D[j]
// M=SEQ, N=Hdim, K=Gdim.  Same software pipelining as GEMM1.
__global__ void __launch_bounds__(256)
gemm2_kernel(const __bf16* __restrict__ Pb,   // SEQ x G
             const __bf16* __restrict__ Cb,   // H x G  (row-major) == B-operand (N,K)
             const float*  __restrict__ x,    // SEQ x H
             const float*  __restrict__ D,    // H
             float* __restrict__ out) {       // SEQ x H
    const int lane  = threadIdx.x & 31;
    const int wave  = threadIdx.x >> 5;
    const int mtile = blockIdx.y * 128 + wave * 16;
    const int ntile = blockIdx.x * 64;

    v8f acc0 = {}, acc1 = {}, acc2 = {}, acc3 = {};

    v16bf a  = load_a_frag(Pb, Gdim, mtile, 0, lane);
    v16bf b0 = load_b_frag(Cb, Gdim, ntile +  0, 0, lane);
    v16bf b1 = load_b_frag(Cb, Gdim, ntile + 16, 0, lane);
    v16bf b2 = load_b_frag(Cb, Gdim, ntile + 32, 0, lane);
    v16bf b3 = load_b_frag(Cb, Gdim, ntile + 48, 0, lane);

#pragma unroll 2
    for (int k0 = 0; k0 < Gdim - 32; k0 += 32) {
        v16bf an  = load_a_frag(Pb, Gdim, mtile, k0 + 32, lane);
        v16bf bn0 = load_b_frag(Cb, Gdim, ntile +  0, k0 + 32, lane);
        v16bf bn1 = load_b_frag(Cb, Gdim, ntile + 16, k0 + 32, lane);
        v16bf bn2 = load_b_frag(Cb, Gdim, ntile + 32, k0 + 32, lane);
        v16bf bn3 = load_b_frag(Cb, Gdim, ntile + 48, k0 + 32, lane);
        acc0 = WMMA_BF16(a, b0, acc0);
        acc1 = WMMA_BF16(a, b1, acc1);
        acc2 = WMMA_BF16(a, b2, acc2);
        acc3 = WMMA_BF16(a, b3, acc3);
        a = an; b0 = bn0; b1 = bn1; b2 = bn2; b3 = bn3;
    }
    acc0 = WMMA_BF16(a, b0, acc0);
    acc1 = WMMA_BF16(a, b1, acc1);
    acc2 = WMMA_BF16(a, b2, acc2);
    acc3 = WMMA_BF16(a, b3, acc3);

    const int ncol  = lane & 15;
    const int mbase = mtile + (lane >> 4) * 8;
#pragma unroll
    for (int j = 0; j < 8; ++j) {
        int m = mbase + j;
        size_t r = (size_t)m * Hdim + ntile;
#pragma unroll
        for (int q = 0; q < 4; ++q) {
            int n = ncol + q * 16;
            float acc = (q == 0) ? acc0[j] : (q == 1) ? acc1[j] : (q == 2) ? acc2[j] : acc3[j];
            out[r + n] = acc + x[r + n] * D[ntile + n];
        }
    }
}

// ---------------- launch ----------------

extern "C" void kernel_launch(void* const* d_in, const int* in_sizes, int n_in,
                              void* d_out, int out_size, void* d_ws, size_t ws_size,
                              hipStream_t stream) {
    const float* x       = (const float*)d_in[0];   // SEQ x H
    const float* c       = (const float*)d_in[1];   // G
    const float* kp_diag = (const float*)d_in[2];   // 32
    const float* k_diag  = (const float*)d_in[3];   // 32
    const float* dtm     = (const float*)d_in[4];   // 32x32
    const float* Bm      = (const float*)d_in[5];   // G x H
    const float* Cm      = (const float*)d_in[6];   // H x G
    const float* Dv      = (const float*)d_in[7];   // H
    float* out = (float*)d_out;

    char* ws = (char*)d_ws;
    float* scaling = (float*)(ws);                       // 4 KB
    float* lr      = scaling + Gdim;
    float* li      = lr + Gdim;
    float* Ea      = (float*)(ws + (64 << 10));          // 4 x 256 KB
    float* Eb      = Ea + NCH * Gdim;
    float* Sa      = Eb + NCH * Gdim;
    float* Sb      = Sa + NCH * Gdim;
    unsigned short* Xb = (unsigned short*)(ws + (2u  << 20));  // 8 MB
    unsigned short* Bb = (unsigned short*)(ws + (10u << 20));  // 1 MB
    unsigned short* Cb = (unsigned short*)(ws + (11u << 20));  // 1 MB
    float*          U  = (float*)(ws + (12u << 20));           // 32 MB
    unsigned short* P  = (unsigned short*)(ws + (44u << 20));  // 16 MB

    // 1) f32 -> bf16 conversions
    cvt_bf16_kernel<<<(SEQ * Hdim / 4 + 255) / 256, 256, 0, stream>>>(x,  Xb, SEQ * Hdim);
    cvt_bf16_kernel<<<(Gdim * Hdim / 4 + 255) / 256, 256, 0, stream>>>(Bm, Bb, Gdim * Hdim);
    cvt_bf16_kernel<<<(Hdim * Gdim / 4 + 255) / 256, 256, 0, stream>>>(Cm, Cb, Hdim * Gdim);

    // 2) per-channel parameters
    params_kernel<<<Gdim / 256, 256, 0, stream>>>(c, kp_diag, k_diag, dtm, scaling, lr, li);

    // 3) GEMM1 (bf16 WMMA) + fused scan pass 1
    gemm1_kernel<<<dim3(Gdim / 64, SEQ / 128), 256, 0, stream>>>(
        (const __bf16*)Xb, (const __bf16*)Bb, scaling, lr, li, U, Ea, Eb);

    // 4) carry propagation + replay -> P = 2*Re(m2) in bf16
    scan_pass2<<<Gdim / 256, 256, 0, stream>>>(lr, li, Ea, Eb, Sa, Sb);
    scan_pass3<<<dim3(Gdim / 256, NCH), 256, 0, stream>>>(U, lr, li, Sa, Sb, P);

    // 5) GEMM2 (bf16 WMMA): out = P @ C.T + x * D
    gemm2_kernel<<<dim3(Hdim / 64, SEQ / 128), 256, 0, stream>>>(
        (const __bf16*)P, (const __bf16*)Cb, x, Dv, out);
}
